// CohereAttention_19679540150759
// MI455X (gfx1250) — matrix operand order repaired and checked by reading, not compiled, not run on hardware
//
#include <hip/hip_runtime.h>
#include <stdint.h>

// ---------------------------------------------------------------- types
typedef __attribute__((ext_vector_type(16))) __bf16    v16bf;
typedef __attribute__((ext_vector_type(8)))  float     v8f;
typedef __attribute__((ext_vector_type(4)))  uint32_t  v4u;
typedef __attribute__((ext_vector_type(2)))  uint32_t  v2u;
typedef unsigned short u16;

#define T_TOK 2048
#define H_DIM 4096
#define NHQ   32
#define NKV   8
#define HD    128
#define QKV_N ((NHQ + 2 * NKV) * HD)   // 6144
#define REP   (NHQ / NKV)              // 4

__device__ __forceinline__ u16 f2bf(float f) {
  union { float f; uint32_t u; } x; x.f = f;
  uint32_t r = x.u + 0x7FFFu + ((x.u >> 16) & 1u);   // round-to-nearest-even
  return (u16)(r >> 16);
}

__device__ __forceinline__ v8f v8f_zero() {
  v8f z = {0.f, 0.f, 0.f, 0.f, 0.f, 0.f, 0.f, 0.f};
  return z;
}

// LDS aperture: generic address low 32 bits == LDS byte offset (ISA 10.2)
__device__ __forceinline__ uint32_t lds_off(const void* p) {
  return (uint32_t)(uintptr_t)p;
}

// CDNA5 async global->LDS copy, 16 bytes per lane, tracked by ASYNCcnt.
__device__ __forceinline__ void async_b128(uint32_t lds_addr, const void* gaddr) {
  asm volatile("global_load_async_to_lds_b128 %0, %1, off"
               :: "v"(lds_addr), "v"(gaddr) : "memory");
}
__device__ __forceinline__ void wait_async0() {
  asm volatile("s_wait_asynccnt 0x0" ::: "memory");
}

// bf16 fragment: 16 elements = 32 bytes = two 16B chunks
union FragBF {
  v16bf v;
  v4u   q[2];
  u16   e[16];
};

// =====================================================================
// Tiled GEMM: C[M,N](f32) = A[M,K] x B[K,N](f32), bf16 WMMA math.
// Block = 256 threads (8 waves), tile 128x128, K-step 32.
// Wave grid 4(M) x 2(N): each wave owns 32x64 = 2x4 WMMA tiles.
// A path: bf16 input -> async global->LDS DMA; f32 input -> convert.
// =====================================================================
template <bool ABF16>
__global__ __launch_bounds__(256) void gemm_bf16_wmma(
    const void* __restrict__ Ap, const float* __restrict__ B,
    float* __restrict__ C, int M, int N, int K) {
  __shared__ __align__(16) u16 As[128 * 32];   // [m][k]
  __shared__ __align__(16) u16 Bs[128 * 32];   // transposed: [n][k]

  const int tid  = threadIdx.x;
  const int lane = tid & 31;
  const int wid  = tid >> 5;
  const int wm   = wid & 3;          // 0..3 -> m offset 32*wm
  const int wn   = wid >> 2;         // 0..1 -> n offset 64*wn
  const int m0   = blockIdx.y * 128;
  const int n0   = blockIdx.x * 128;

  v8f acc[2][4];
#pragma unroll
  for (int i = 0; i < 2; ++i)
#pragma unroll
    for (int j = 0; j < 4; ++j) acc[i][j] = v8f_zero();

  const int ar = tid >> 1, ac = (tid & 1) * 16;   // A loader: 128 rows x 32 cols
  const int br = tid >> 3, bc = (tid & 7) * 16;   // B loader: 32 rows x 128 cols
  const int ln = lane & 15;
  const int hh = (lane >> 4) * 8;    // A-frag K sub-offset
  const int kh = (lane >> 4) * 16;   // unused (B frags also use hh)
  const uint32_t lds_a = lds_off(&As[ar * 32 + ac]);

  for (int k0 = 0; k0 < K; k0 += 32) {
    // ---- A tile -> LDS
    if (ABF16) {
      // straight bf16 copy: async DMA, bypasses VGPRs
      const u16* Ab = (const u16*)Ap + (size_t)(m0 + ar) * K + k0 + ac;
      async_b128(lds_a,      Ab);
      async_b128(lds_a + 16, Ab + 8);
    } else {
      const float* Af = (const float*)Ap + (size_t)(m0 + ar) * K + k0 + ac;
#pragma unroll
      for (int i = 0; i < 16; i += 4) {
        float4 f = *(const float4*)(Af + i);
        As[ar * 32 + ac + i + 0] = f2bf(f.x);
        As[ar * 32 + ac + i + 1] = f2bf(f.y);
        As[ar * 32 + ac + i + 2] = f2bf(f.z);
        As[ar * 32 + ac + i + 3] = f2bf(f.w);
      }
    }
    // ---- B tile -> LDS transposed [n][k] (f32 -> bf16 convert)
    {
      const float* Bf = B + (size_t)(k0 + br) * N + n0 + bc;
#pragma unroll
      for (int i = 0; i < 16; i += 4) {
        float4 f = *(const float4*)(Bf + i);
        Bs[(bc + i + 0) * 32 + br] = f2bf(f.x);
        Bs[(bc + i + 1) * 32 + br] = f2bf(f.y);
        Bs[(bc + i + 2) * 32 + br] = f2bf(f.z);
        Bs[(bc + i + 3) * 32 + br] = f2bf(f.w);
      }
      if (k0 + 32 < K)   // gfx1250 global_prefetch of next B tile
        __builtin_prefetch(B + (size_t)(k0 + 32 + br) * N + n0 + bc, 0, 0);
    }
    if (ABF16) wait_async0();
    __syncthreads();

    // ---- fragments + 8 WMMAs
    FragBF a[2], b[4];
#pragma unroll
    for (int mi = 0; mi < 2; ++mi) {
      int row = wm * 32 + mi * 16 + ln;
      a[mi].q[0] = *(const v4u*)&As[row * 32 + hh];
      a[mi].q[1] = *(const v4u*)&As[row * 32 + hh + 16];
    }
#pragma unroll
    for (int ni = 0; ni < 4; ++ni) {
      int n = wn * 64 + ni * 16 + ln;
      b[ni].q[0] = *(const v4u*)&Bs[n * 32 + hh];
      b[ni].q[1] = *(const v4u*)&Bs[n * 32 + hh + 16];
    }
#pragma unroll
    for (int mi = 0; mi < 2; ++mi)
#pragma unroll
      for (int ni = 0; ni < 4; ++ni)
        acc[mi][ni] = __builtin_amdgcn_wmma_f32_16x16x32_bf16(
            false, a[mi].v, false, b[ni].v, (short)0, acc[mi][ni], false, false);
    __syncthreads();
  }

  // ---- epilogue: C/D layout: lane -> col = lane&15, row = vgpr + 8*(lane>=16)
  const int cn  = n0 + wn * 64 + ln;
  const int cmb = m0 + wm * 32 + ((lane >> 4) << 3);
#pragma unroll
  for (int mi = 0; mi < 2; ++mi)
#pragma unroll
    for (int ni = 0; ni < 4; ++ni)
#pragma unroll
      for (int r = 0; r < 8; ++r)
        C[(size_t)(cmb + mi * 16 + r) * N + cn + ni * 16] = acc[mi][ni][r];
}

// =====================================================================
// Per-head LayerNorm + interleaved RoPE + bf16 pack.
// One block per token (8 waves); one wave handles one head (HD=128 ->
// 4 f32 per lane in wave32). Q gets HD^-0.5 folded in.
// =====================================================================
__global__ __launch_bounds__(256) void norm_rope_kernel(
    const int* __restrict__ positions, const float* __restrict__ qkv,
    const float* __restrict__ qw, const float* __restrict__ kw,
    u16* __restrict__ qbf, u16* __restrict__ kbf, u16* __restrict__ vbf) {
  const int t    = blockIdx.x;
  const int tid  = threadIdx.x;
  const int lane = tid & 31;
  const int wid  = tid >> 5;
  const float pos = (float)positions[t];
  const float* row = qkv + (size_t)t * QKV_N;

  // RoPE angles for this lane's two pairs (pair indices 2*lane, 2*lane+1)
  const float i0 = (float)(2 * lane), i1 = (float)(2 * lane + 1);
  float c0, s0, c1, s1;
  {
    float f0 = pos * __powf(10000.f, -i0 * (1.f / 64.f));
    float f1 = pos * __powf(10000.f, -i1 * (1.f / 64.f));
    __sincosf(f0, &s0, &c0);
    __sincosf(f1, &s1, &c1);
  }
  const float qscale = 0.0883883476483184f;   // 128^-0.5

  for (int idx = wid; idx < NHQ + NKV; idx += 8) {
    const bool is_q = idx < NHQ;
    const int  hh   = is_q ? idx : idx - NHQ;
    const float* src = is_q ? (row + hh * HD) : (row + NHQ * HD + hh * HD);
    float4 x = *(const float4*)(src + lane * 4);
    float s  = x.x + x.y + x.z + x.w;
    float ss = x.x * x.x + x.y * x.y + x.z * x.z + x.w * x.w;
#pragma unroll
    for (int m = 1; m < 32; m <<= 1) {
      s  += __shfl_xor(s, m, 32);
      ss += __shfl_xor(ss, m, 32);
    }
    float mean = s * (1.f / 128.f);
    float var  = ss * (1.f / 128.f) - mean * mean;
    float inv  = rsqrtf(var + 1e-5f);
    const float* w = (is_q ? qw : kw) + hh * HD + lane * 4;
    float y0 = (x.x - mean) * inv * w[0];
    float y1 = (x.y - mean) * inv * w[1];
    float y2 = (x.z - mean) * inv * w[2];
    float y3 = (x.w - mean) * inv * w[3];
    float o0 = y0 * c0 - y1 * s0;
    float o1 = y1 * c0 + y0 * s0;
    float o2 = y2 * c1 - y3 * s1;
    float o3 = y3 * c1 + y2 * s1;
    if (is_q) { o0 *= qscale; o1 *= qscale; o2 *= qscale; o3 *= qscale; }
    u16* dst = is_q ? (qbf + ((size_t)t * NHQ + hh) * HD + lane * 4)
                    : (kbf + ((size_t)t * NKV + hh) * HD + lane * 4);
    v2u pk;
    pk.x = (uint32_t)f2bf(o0) | ((uint32_t)f2bf(o1) << 16);
    pk.y = (uint32_t)f2bf(o2) | ((uint32_t)f2bf(o3) << 16);
    *(v2u*)dst = pk;
  }

  // V: plain f32 -> bf16 convert (layout [t][kv][d] matches qkv tail)
  {
    const float* vsrc = row + (NHQ + NKV) * HD;
    float4 f = *(const float4*)(vsrc + tid * 4);
    v2u pk;
    pk.x = (uint32_t)f2bf(f.x) | ((uint32_t)f2bf(f.y) << 16);
    pk.y = (uint32_t)f2bf(f.z) | ((uint32_t)f2bf(f.w) << 16);
    *(v2u*)(vbf + (size_t)t * NKV * HD + tid * 4) = pk;
  }
}

// =====================================================================
// Flash attention, causal, GQA (kv head = h / REP).
// Block: 1 head x 128 q rows (8 waves, 16 rows/wave). KV step = 32.
// Q A-fragments held in registers; K tile streamed via async DMA.
// =====================================================================
__global__ __launch_bounds__(256) void flash_kernel(
    const u16* __restrict__ qbf, const u16* __restrict__ kbf,
    const u16* __restrict__ vbf, u16* __restrict__ obf) {
  __shared__ __align__(16) u16 Ks[32 * 128];       // [kv][d]
  __shared__ __align__(16) u16 Vt[128 * 32];       // transposed [d][kv]
  __shared__ __align__(16) u16 Ps[8 * 16 * 32];    // per-wave P staging

  const int tid  = threadIdx.x;
  const int lane = tid & 31;
  const int wid  = tid >> 5;
  const int qb   = blockIdx.x;
  const int h    = blockIdx.y;
  const int kvh  = h / REP;
  const int hi   = lane >> 4;     // half-wave
  const int ln   = lane & 15;
  const int hh   = hi * 8;        // A-frag K sub-offset
  const int kh   = hi * 16;       // unused (B frags also use hh)

  // Q fragments (A layout), kept in registers
  FragBF qf[4];
  {
    const int  qr   = qb * 128 + wid * 16 + ln;
    const u16* qrow = qbf + ((size_t)qr * NHQ + h) * HD;
#pragma unroll
    for (int kk = 0; kk < 4; ++kk) {
      qf[kk].q[0] = *(const v4u*)(qrow + kk * 32 + hh);
      qf[kk].q[1] = *(const v4u*)(qrow + kk * 32 + hh + 16);
    }
  }

  v8f  O[8];
  float mrow[8], lrow[8];
#pragma unroll
  for (int i = 0; i < 8; ++i) { O[i] = v8f_zero(); mrow[i] = -1e30f; lrow[i] = 0.f; }

  const int nkb = (qb + 1) * 4;              // causal: kv up to end of q block
  const int kr = tid >> 3, kc = (tid & 7) * 16;
  const uint32_t lds_k = lds_off(&Ks[kr * 128 + kc]);

  for (int kb = 0; kb < nkb; ++kb) {
    // ---- K tile: async global->LDS DMA (bf16, layout-preserving)
    {
      const size_t kvrow = (size_t)(kb * 32 + kr) * NKV + kvh;
      const u16* ks = kbf + kvrow * HD + kc;
      async_b128(lds_k,      ks);
      async_b128(lds_k + 16, ks + 8);
      // ---- V tile: needs transpose -> register path
      const u16* vs = vbf + kvrow * HD + kc;
      FragBF tmp;
      tmp.q[0] = *(const v4u*)vs;
      tmp.q[1] = *(const v4u*)(vs + 8);
#pragma unroll
      for (int i = 0; i < 16; ++i) Vt[(kc + i) * 32 + kr] = tmp.e[i];
    }
    wait_async0();
    __syncthreads();

    // ---- S = Q x K^T : two 16x16 tiles (kv cols 0-15, 16-31)
    v8f sc0 = v8f_zero(), sc1 = v8f_zero();
#pragma unroll
    for (int kk = 0; kk < 4; ++kk) {
      FragBF b0, b1;
      b0.q[0] = *(const v4u*)&Ks[(ln)      * 128 + kk * 32 + hh];
      b0.q[1] = *(const v4u*)&Ks[(ln)      * 128 + kk * 32 + hh + 16];
      b1.q[0] = *(const v4u*)&Ks[(16 + ln) * 128 + kk * 32 + hh];
      b1.q[1] = *(const v4u*)&Ks[(16 + ln) * 128 + kk * 32 + hh + 16];
      sc0 = __builtin_amdgcn_wmma_f32_16x16x32_bf16(false, qf[kk].v, false, b0.v,
                                                    (short)0, sc0, false, false);
      sc1 = __builtin_amdgcn_wmma_f32_16x16x32_bf16(false, qf[kk].v, false, b1.v,
                                                    (short)0, sc1, false, false);
    }

    // ---- causal mask + online softmax (rows live per-VGPR in half-wave)
    const int qrow0 = qb * 128 + wid * 16 + hi * 8;
    const int col0  = kb * 32 + ln;
    const int col1  = col0 + 16;
    float alpha[8];
#pragma unroll
    for (int r = 0; r < 8; ++r) {
      const int qr = qrow0 + r;
      float v0 = (col0 <= qr) ? sc0[r] : -1e30f;
      float v1 = (col1 <= qr) ? sc1[r] : -1e30f;
      float mx = fmaxf(v0, v1);
      mx = fmaxf(mx, __shfl_xor(mx, 1, 32));
      mx = fmaxf(mx, __shfl_xor(mx, 2, 32));
      mx = fmaxf(mx, __shfl_xor(mx, 4, 32));
      mx = fmaxf(mx, __shfl_xor(mx, 8, 32));
      float mnew = fmaxf(mrow[r], mx);
      alpha[r]   = __expf(mrow[r] - mnew);
      mrow[r]    = mnew;
      float p0 = __expf(v0 - mnew);
      float p1 = __expf(v1 - mnew);
      float sum = p0 + p1;
      sum += __shfl_xor(sum, 1, 32);
      sum += __shfl_xor(sum, 2, 32);
      sum += __shfl_xor(sum, 4, 32);
      sum += __shfl_xor(sum, 8, 32);
      lrow[r] = lrow[r] * alpha[r] + sum;
      // stage P (C layout -> LDS) for A-layout reload
      Ps[wid * 512 + (r + hi * 8) * 32 + ln]      = f2bf(p0);
      Ps[wid * 512 + (r + hi * 8) * 32 + 16 + ln] = f2bf(p1);
    }
#pragma unroll
    for (int nt = 0; nt < 8; ++nt)
#pragma unroll
      for (int r = 0; r < 8; ++r) O[nt][r] *= alpha[r];

    // wave-local LDS write->read ordering (CDNA5 split counters)
    asm volatile("s_wait_dscnt 0" ::: "memory");

    // ---- O += P x V
    FragBF pa;
    pa.q[0] = *(const v4u*)&Ps[wid * 512 + ln * 32 + hh];
    pa.q[1] = *(const v4u*)&Ps[wid * 512 + ln * 32 + hh + 16];
#pragma unroll
    for (int nt = 0; nt < 8; ++nt) {
      FragBF bv;
      bv.q[0] = *(const v4u*)&Vt[(nt * 16 + ln) * 32 + hh];
      bv.q[1] = *(const v4u*)&Vt[(nt * 16 + ln) * 32 + hh + 16];
      O[nt] = __builtin_amdgcn_wmma_f32_16x16x32_bf16(false, pa.v, false, bv.v,
                                                      (short)0, O[nt], false, false);
    }
    __syncthreads();
  }

  // ---- finalize: divide by row sums, emit bf16 attn output [q][h*HD+d]
  const int qrow0 = qb * 128 + wid * 16 + hi * 8;
#pragma unroll
  for (int r = 0; r < 8; ++r) {
    const float inv = 1.f / lrow[r];
#pragma unroll
    for (int nt = 0; nt < 8; ++nt)
      obf[(size_t)(qrow0 + r) * H_DIM + h * HD + nt * 16 + ln] = f2bf(O[nt][r] * inv);
  }
}

// =====================================================================
extern "C" void kernel_launch(void* const* d_in, const int* in_sizes, int n_in,
                              void* d_out, int out_size, void* d_ws, size_t ws_size,
                              hipStream_t stream) {
  (void)in_sizes; (void)n_in; (void)out_size; (void)ws_size;
  const int*   positions = (const int*)d_in[0];
  const float* hidden    = (const float*)d_in[1];
  const float* w_qkv     = (const float*)d_in[2];
  const float* w_o       = (const float*)d_in[3];
  const float* q_norm_w  = (const float*)d_in[4];
  const float* k_norm_w  = (const float*)d_in[5];
  float*       out       = (float*)d_out;

  char*  ws  = (char*)d_ws;
  size_t off = 0;
  auto alloc = [&](size_t bytes) -> void* {
    off = (off + 255) & ~(size_t)255;
    void* p = ws + off;
    off += bytes;
    return p;
  };
  float* qkv    = (float*)alloc((size_t)T_TOK * QKV_N * sizeof(float));  // 48 MiB
  u16*   qbf    = (u16*)alloc((size_t)T_TOK * NHQ * HD * 2);             // 16 MiB
  u16*   kbf    = (u16*)alloc((size_t)T_TOK * NKV * HD * 2);             //  4 MiB
  u16*   vbf    = (u16*)alloc((size_t)T_TOK * NKV * HD * 2);             //  4 MiB
  u16*   attnbf = (u16*)alloc((size_t)T_TOK * NHQ * HD * 2);             // 16 MiB

  // 1) QKV projection
  gemm_bf16_wmma<false><<<dim3(QKV_N / 128, T_TOK / 128), 256, 0, stream>>>(
      hidden, w_qkv, qkv, T_TOK, QKV_N, H_DIM);
  // 2) per-head LayerNorm + RoPE + bf16 pack (q pre-scaled by HD^-0.5)
  norm_rope_kernel<<<T_TOK, 256, 0, stream>>>(positions, qkv, q_norm_w, k_norm_w,
                                              qbf, kbf, vbf);
  // 3) causal GQA flash attention (async K streaming)
  flash_kernel<<<dim3(T_TOK / 128, NHQ), 256, 0, stream>>>(qbf, kbf, vbf, attnbf);
  // 4) output projection (async A streaming, bf16 A)
  gemm_bf16_wmma<true><<<dim3(H_DIM / 128, T_TOK / 128), 256, 0, stream>>>(
      attnbf, w_o, out, T_TOK, H_DIM, H_DIM);
}
